// STGNN_GhostFusor_BG_AR_87471303950930
// MI455X (gfx1250) — matrix operand order repaired and checked by reading, not compiled, not run on hardware
//
#include <hip/hip_runtime.h>
#include <hip/hip_bf16.h>
#include <math.h>

// ---------------- problem constants (match reference) ----------------
constexpr int NN     = 20000;   // nodes
constexpr int NGHOST = 2000;
constexpr int T_IN   = 8;
constexpr int FEAT   = 16;
constexpr int HID    = 128;
constexpr int NE     = 320000;  // edges
constexpr int H3     = 384;
constexpr float ALPHA = 0.5f;

typedef __attribute__((ext_vector_type(16))) _Float16 v16h;
typedef __attribute__((ext_vector_type(8)))  float    v8f;

// ---------------- WMMA helpers ----------------
// A-fragment (16x32 f16 tile) per ISA layout: lanes 0-15 rows 0-15 (K base 0),
// lanes 16-31 rows 0-15 (K base +8); VGPR j holds K pairs.
__device__ inline v16h load_a_frag(const _Float16* At, int lane, int kt, int kpad) {
    v16h a;
    int row = lane & 15;
    int kb  = (lane >> 4) << 3;   // 0 or 8
#pragma unroll
    for (int j = 0; j < 8; ++j) {
        int k0 = kt * 32 + ((j < 4) ? 0 : 16) + kb + ((j & 3) << 1);
        a[2 * j]     = At[row * kpad + k0];
        a[2 * j + 1] = At[row * kpad + k0 + 1];
    }
    return a;
}

__device__ inline float sigmoidf_(float x) { return 1.0f / (1.0f + __expf(-x)); }

// ---------------- weight pre-fragmentation (f32 -> f16 B-fragments) ----------------
// B (32x16) layout: lane holds column n = lane&15; lanes>=16 carry K+8; elements 8-15 carry K+16.
__global__ void prep_weights_kernel(const float* __restrict__ W, _Float16* __restrict__ frag,
                                    int KV, int KT, int NT) {
    int idx = blockIdx.x * 256 + threadIdx.x;
    int total = NT * KT * 32 * 16;
    if (idx >= total) return;
    int e    = idx & 15;
    int lane = (idx >> 4) & 31;
    int kt   = (idx >> 9) % KT;
    int nt   = idx / (512 * KT);
    int k = kt * 32 + ((e < 8) ? 0 : 16) + ((lane >= 16) ? 8 : 0) + (e & 7);
    int n = nt * 16 + (lane & 15);
    float v = (k < KV) ? W[(size_t)k * (NT * 16) + n] : 0.0f;
    frag[idx] = (_Float16)v;
}

// ---------------- generic WMMA GEMM: C[M,NT*16] = act(A[M,KV] @ W + bias) ----------------
template <int KV, int KPAD, int NT, bool RELU>
__global__ __launch_bounds__(128) void wmma_gemm_kernel(const float* __restrict__ A,
                                                        const _Float16* __restrict__ Wfrag,
                                                        const float* __restrict__ bias,
                                                        float* __restrict__ C) {
    constexpr int KT   = KPAD / 32;
    constexpr int NCOL = NT * 16;
    __shared__ _Float16 At[16 * KPAD];
    int tid  = threadIdx.x;
    int mblk = blockIdx.x;
    for (int idx = tid; idx < 16 * KPAD; idx += 128) {
        int r = idx / KPAD, k = idx % KPAD;
        float v = (k < KV) ? A[(size_t)(mblk * 16 + r) * KV + k] : 0.0f;
        At[idx] = (_Float16)v;
    }
    __syncthreads();
    int lane = tid & 31, wave = tid >> 5;
    v16h af[KT];
#pragma unroll
    for (int kt = 0; kt < KT; ++kt) af[kt] = load_a_frag(At, lane, kt, KPAD);

    for (int nt = wave; nt < NT; nt += 4) {
        v8f acc = {};
#pragma unroll
        for (int kt = 0; kt < KT; ++kt) {
            v16h bf = *(const v16h*)(Wfrag + (((size_t)nt * KT + kt) * 32 + lane) * 16);
            acc = __builtin_amdgcn_wmma_f32_16x16x32_f16(false, af[kt], false, bf,
                                                         (short)0, acc, false, false);
        }
        int col   = nt * 16 + (lane & 15);
        int rbase = mblk * 16 + ((lane >> 4) << 3);
        float bb  = bias ? bias[col] : 0.0f;
#pragma unroll
        for (int r = 0; r < 8; ++r) {
            float v = acc[r] + bb;
            if (RELU) v = fmaxf(v, 0.0f);
            C[(size_t)(rbase + r) * NCOL + col] = v;
        }
    }
}

// ---------------- fused GRU: gi = g2@W_ih, gh = h@W_hh, gate math, h update ----------------
__global__ __launch_bounds__(128) void gru_fused_kernel(const float* __restrict__ g2,
                                                        const _Float16* __restrict__ fWih,
                                                        const _Float16* __restrict__ fWhh,
                                                        const float* __restrict__ b_ih,
                                                        const float* __restrict__ b_hh,
                                                        float* __restrict__ h) {
    __shared__ _Float16 Gt[16 * 128];
    __shared__ _Float16 Ht[16 * 128];
    int tid = threadIdx.x, mblk = blockIdx.x;
    for (int idx = tid; idx < 2048; idx += 128) {
        int r = idx >> 7, k = idx & 127;
        size_t off = (size_t)(mblk * 16 + r) * 128 + k;
        Gt[idx] = (_Float16)g2[off];
        Ht[idx] = (_Float16)h[off];
    }
    __syncthreads();
    int lane = tid & 31, wave = tid >> 5;
    v16h ag[4], ah[4];
#pragma unroll
    for (int kt = 0; kt < 4; ++kt) {
        ag[kt] = load_a_frag(Gt, lane, kt, 128);
        ah[kt] = load_a_frag(Ht, lane, kt, 128);
    }
    v8f gi[6], gh[6];
#pragma unroll
    for (int j = 0; j < 6; ++j) {
        int nt = wave + 4 * j;
        v8f ai = {}, ah2 = {};
#pragma unroll
        for (int kt = 0; kt < 4; ++kt) {
            v16h bi = *(const v16h*)(fWih + (((size_t)nt * 4 + kt) * 32 + lane) * 16);
            ai = __builtin_amdgcn_wmma_f32_16x16x32_f16(false, ag[kt], false, bi,
                                                        (short)0, ai, false, false);
            v16h bh = *(const v16h*)(fWhh + (((size_t)nt * 4 + kt) * 32 + lane) * 16);
            ah2 = __builtin_amdgcn_wmma_f32_16x16x32_f16(false, ah[kt], false, bh,
                                                         (short)0, ah2, false, false);
        }
        gi[j] = ai; gh[j] = ah2;
    }
    int colb   = lane & 15;
    int rowoff = (lane >> 4) << 3;
#pragma unroll
    for (int jr = 0; jr < 2; ++jr) {
        int ntr = wave + 4 * jr;
        int col = ntr * 16 + colb;
        float bir = b_ih[col],       bhr = b_hh[col];
        float biz = b_ih[col + 128], bhz = b_hh[col + 128];
        float bin = b_ih[col + 256], bhn = b_hh[col + 256];
#pragma unroll
        for (int r = 0; r < 8; ++r) {
            float rr = sigmoidf_((gi[jr][r] + bir) + (gh[jr][r] + bhr));
            float zz = sigmoidf_((gi[jr + 2][r] + biz) + (gh[jr + 2][r] + bhz));
            float nn = tanhf((gi[jr + 4][r] + bin) + rr * (gh[jr + 4][r] + bhn));
            size_t off = (size_t)(mblk * 16 + rowoff + r) * 128 + col;
            float ho = h[off];
            h[off] = (1.0f - zz) * nn + zz * ho;
        }
    }
}

// ---------------- edge scatter: agg[dst] += hw[src] * norm_e (wave per edge) ----------------
__global__ __launch_bounds__(256) void scatter_kernel(const int* __restrict__ src,
                                                      const int* __restrict__ dst,
                                                      const float* __restrict__ norm_e,
                                                      const float* __restrict__ hw,
                                                      float* __restrict__ agg) {
    int e    = blockIdx.x * 8 + (threadIdx.x >> 5);
    int lane = threadIdx.x & 31;
    if (e >= NE) return;
    int s = src[e], d = dst[e];
    float w = norm_e[e];
    float4 m = ((const float4*)(hw + (size_t)s * HID))[lane];
    float* ap = agg + (size_t)d * HID + lane * 4;
    __hip_atomic_fetch_add(ap + 0, m.x * w, __ATOMIC_RELAXED, __HIP_MEMORY_SCOPE_AGENT);
    __hip_atomic_fetch_add(ap + 1, m.y * w, __ATOMIC_RELAXED, __HIP_MEMORY_SCOPE_AGENT);
    __hip_atomic_fetch_add(ap + 2, m.z * w, __ATOMIC_RELAXED, __HIP_MEMORY_SCOPE_AGENT);
    __hip_atomic_fetch_add(ap + 3, m.w * w, __ATOMIC_RELAXED, __HIP_MEMORY_SCOPE_AGENT);
}

// g = relu(agg + hw*norm_self + bias)
__global__ void gcn_finish_kernel(const float* __restrict__ hw, const float* __restrict__ agg,
                                  const float* __restrict__ nself, const float* __restrict__ bias,
                                  float* __restrict__ g) {
    int i = blockIdx.x * 256 + threadIdx.x;
    if (i >= NN * HID) return;
    int node = i >> 7, c = i & 127;
    g[i] = fmaxf(agg[i] + hw[i] * nself[node] + bias[c], 0.0f);
}

// ---------------- per-step ghost handling ----------------
__global__ void copy_xt_kernel(const float* __restrict__ x, int t, float* __restrict__ xt) {
    int i = blockIdx.x * 256 + threadIdx.x;
    if (i >= NN * FEAT) return;
    int node = i >> 4, f = i & 15;
    xt[i] = x[(size_t)node * T_IN * FEAT + t * FEAT + f];
}

__global__ __launch_bounds__(160) void ghost_fuse_kernel(
    const float* __restrict__ x, int t, const int* __restrict__ gidx,
    const int* __restrict__ gd1, const int* __restrict__ gd2, const float* __restrict__ gdx,
    const float* __restrict__ Wf1, const float* __restrict__ bf1,
    const float* __restrict__ Wf2, const float* __restrict__ bf2,
    const float* __restrict__ Wg1, const float* __restrict__ bg1,
    const float* __restrict__ Wg2, const float* __restrict__ bg2,
    float* __restrict__ xt) {
    __shared__ float feats[49];
    __shared__ float zf[96];
    __shared__ float ghid[48];
    __shared__ float delta[16];
    __shared__ float gate;
    int g = blockIdx.x, tid = threadIdx.x;
    int node = gidx[g];
    if (tid < 16) {
        int n1 = gd1[g], n2 = gd2[g];
        feats[tid]      = x[(size_t)node * T_IN * FEAT + t * FEAT + tid];
        feats[16 + tid] = x[(size_t)n1 * T_IN * FEAT + t * FEAT + tid];
        feats[32 + tid] = x[(size_t)n2 * T_IN * FEAT + t * FEAT + tid];
        if (tid == 0) feats[48] = gdx[g];
    }
    __syncthreads();
    if (tid < 96) {
        float s = bf1[tid];
        for (int k = 0; k < 49; ++k) s += feats[k] * Wf1[k * 96 + tid];
        zf[tid] = fmaxf(s, 0.0f);
    } else if (tid < 144) {
        int u = tid - 96;
        float s = bg1[u];
        for (int k = 0; k < 49; ++k) s += feats[k] * Wg1[k * 48 + u];
        ghid[u] = fmaxf(s, 0.0f);
    }
    __syncthreads();
    if (tid < 16) {
        float s = bf2[tid];
        for (int k = 0; k < 96; ++k) s += zf[k] * Wf2[k * 16 + tid];
        delta[tid] = s;
    } else if (tid == 159) {
        float s = bg2[0];
        for (int k = 0; k < 48; ++k) s += ghid[k] * Wg2[k];
        gate = sigmoidf_(s);
    }
    __syncthreads();
    if (tid < 16) xt[(size_t)node * FEAT + tid] = feats[tid] + ALPHA * delta[tid] * gate;
}

// ---------------- once-per-launch graph preprocessing ----------------
__global__ void init_pre_kernel(int* __restrict__ first_eid, int* __restrict__ deg_i) {
    int i = blockIdx.x * 256 + threadIdx.x;
    if (i >= NN) return;
    first_eid[i] = 0x7FFFFFFF;
    deg_i[i] = 0;
}

__global__ void edge_pre_kernel(const int* __restrict__ src, const int* __restrict__ dst,
                                int* __restrict__ first_eid, int* __restrict__ deg_i) {
    int e = blockIdx.x * 256 + threadIdx.x;
    if (e >= NE) return;
    atomicMin(&first_eid[src[e]], e);
    atomicAdd(&deg_i[dst[e]], 1);
}

__global__ void node_pre_kernel(const int* __restrict__ first_eid, const int* __restrict__ dst,
                                const float* __restrict__ edge_attr, const int* __restrict__ deg_i,
                                int* __restrict__ first_out, float* __restrict__ first_dx,
                                float* __restrict__ dis, float* __restrict__ nself) {
    int i = blockIdx.x * 256 + threadIdx.x;
    if (i >= NN) return;
    int fe = first_eid[i];
    bool has = fe < NE;
    int fes = has ? fe : 0;
    first_out[i] = has ? dst[fes] : -1;
    first_dx[i]  = has ? fmaxf(edge_attr[fes], 1e-6f) : -1.0f;
    float dg = (float)deg_i[i] + 1.0f;
    dis[i]   = rsqrtf(dg);
    nself[i] = 1.0f / dg;
}

__global__ void norm_e_kernel(const int* __restrict__ src, const int* __restrict__ dst,
                              const float* __restrict__ dis, float* __restrict__ norm_e) {
    int e = blockIdx.x * 256 + threadIdx.x;
    if (e >= NE) return;
    norm_e[e] = dis[src[e]] * dis[dst[e]];
}

// ordered compaction of mask==0 node ids (single block, ballot scan)
__global__ __launch_bounds__(1024) void ghost_scan_kernel(const int* __restrict__ mask,
                                                          int* __restrict__ gidx) {
    __shared__ int woff[32];
    __shared__ int base;
    int tid = threadIdx.x, lane = tid & 31, wid = tid >> 5;
    if (tid == 0) base = 0;
    __syncthreads();
    for (int start = 0; start < NN; start += 1024) {
        int i = start + tid;
        int isg = (i < NN) && (mask[i] == 0);
        unsigned b32 = (unsigned)__ballot(isg);
        int prefix = __popc(b32 & ((1u << lane) - 1u));
        if (lane == 0) woff[wid] = __popc(b32);
        __syncthreads();
        if (tid == 0) {
            int s = base;
            for (int w = 0; w < 32; ++w) { int c = woff[w]; woff[w] = s; s += c; }
            base = s;
        }
        __syncthreads();
        if (isg) {
            int pos = woff[wid] + prefix;
            if (pos < NGHOST) gidx[pos] = i;
        }
        __syncthreads();
    }
}

__global__ void ghost_hops_kernel(const int* __restrict__ gidx, const int* __restrict__ first_out,
                                  const float* __restrict__ first_dx, int* __restrict__ gd1,
                                  int* __restrict__ gd2, float* __restrict__ gdx) {
    int g = blockIdx.x * 256 + threadIdx.x;
    if (g >= NGHOST) return;
    int n = gidx[g];
    int b = first_out[n];
    if (b < 0) b = n;
    gdx[g] = fmaxf(first_dx[n], 1e-6f);
    int n1 = first_out[b];  if (n1 < 0) n1 = b;
    int n2 = first_out[n1]; if (n2 < 0) n2 = n1;
    gd1[g] = n1; gd2[g] = n2;
}

// ---------------- host orchestration ----------------
extern "C" void kernel_launch(void* const* d_in, const int* in_sizes, int n_in,
                              void* d_out, int out_size, void* d_ws, size_t ws_size,
                              hipStream_t stream) {
    (void)in_sizes; (void)n_in; (void)out_size; (void)ws_size;
    const float* x     = (const float*)d_in[0];
    const float* eattr = (const float*)d_in[1];
    const float* W_lin = (const float*)d_in[2];
    const float* b_lin = (const float*)d_in[3];
    const float* Wc1   = (const float*)d_in[4];
    const float* bc1   = (const float*)d_in[5];
    const float* Wc2   = (const float*)d_in[6];
    const float* bc2   = (const float*)d_in[7];
    const float* W_ih  = (const float*)d_in[8];
    const float* b_ih  = (const float*)d_in[9];
    const float* W_hh  = (const float*)d_in[10];
    const float* b_hh  = (const float*)d_in[11];
    const float* Wf1   = (const float*)d_in[12];
    const float* bf1   = (const float*)d_in[13];
    const float* Wf2   = (const float*)d_in[14];
    const float* bf2   = (const float*)d_in[15];
    const float* Wg1   = (const float*)d_in[16];
    const float* bg1   = (const float*)d_in[17];
    const float* Wg2   = (const float*)d_in[18];
    const float* bg2   = (const float*)d_in[19];
    const int* eidx    = (const int*)d_in[20];
    const int* mask    = (const int*)d_in[21];
    const int* src = eidx;
    const int* dst = eidx + NE;
    float* h = (float*)d_out;

    char* ws = (char*)d_ws;
    size_t off = 0;
    auto alloc = [&](size_t bytes) -> char* {
        char* p = ws + off;
        off += (bytes + 255) & ~(size_t)255;
        return p;
    };
    _Float16* fWlin = (_Float16*)alloc((size_t)8 * 1 * 512 * 2);
    _Float16* fWc1  = (_Float16*)alloc((size_t)8 * 4 * 512 * 2);
    _Float16* fWc2  = (_Float16*)alloc((size_t)8 * 4 * 512 * 2);
    _Float16* fWih  = (_Float16*)alloc((size_t)24 * 4 * 512 * 2);
    _Float16* fWhh  = (_Float16*)alloc((size_t)24 * 4 * 512 * 2);
    int*   first_eid = (int*)alloc((size_t)NN * 4);
    int*   deg_i     = (int*)alloc((size_t)NN * 4);
    int*   first_out = (int*)alloc((size_t)NN * 4);
    float* first_dx  = (float*)alloc((size_t)NN * 4);
    float* dis       = (float*)alloc((size_t)NN * 4);
    float* nself     = (float*)alloc((size_t)NN * 4);
    float* norm_e    = (float*)alloc((size_t)NE * 4);
    int*   gidx      = (int*)alloc((size_t)NGHOST * 4);
    int*   gd1       = (int*)alloc((size_t)NGHOST * 4);
    int*   gd2       = (int*)alloc((size_t)NGHOST * 4);
    float* gdx       = (float*)alloc((size_t)NGHOST * 4);
    float* xt  = (float*)alloc((size_t)NN * FEAT * 4);
    float* g   = (float*)alloc((size_t)NN * HID * 4);
    float* hw  = (float*)alloc((size_t)NN * HID * 4);
    float* agg = (float*)alloc((size_t)NN * HID * 4);

    const int MBLK = NN / 16;  // 1250, exact

    // h0 = 0
    hipMemsetAsync(d_out, 0, (size_t)NN * HID * 4, stream);

    // weight fragmentation (once)
    prep_weights_kernel<<<(8 * 1 * 512 + 255) / 256, 256, 0, stream>>>(W_lin, fWlin, 16, 1, 8);
    prep_weights_kernel<<<(8 * 4 * 512 + 255) / 256, 256, 0, stream>>>(Wc1, fWc1, 128, 4, 8);
    prep_weights_kernel<<<(8 * 4 * 512 + 255) / 256, 256, 0, stream>>>(Wc2, fWc2, 128, 4, 8);
    prep_weights_kernel<<<(24 * 4 * 512 + 255) / 256, 256, 0, stream>>>(W_ih, fWih, 128, 4, 24);
    prep_weights_kernel<<<(24 * 4 * 512 + 255) / 256, 256, 0, stream>>>(W_hh, fWhh, 128, 4, 24);

    // graph preprocessing (once)
    init_pre_kernel<<<(NN + 255) / 256, 256, 0, stream>>>(first_eid, deg_i);
    edge_pre_kernel<<<(NE + 255) / 256, 256, 0, stream>>>(src, dst, first_eid, deg_i);
    node_pre_kernel<<<(NN + 255) / 256, 256, 0, stream>>>(first_eid, dst, eattr, deg_i,
                                                          first_out, first_dx, dis, nself);
    norm_e_kernel<<<(NE + 255) / 256, 256, 0, stream>>>(src, dst, dis, norm_e);
    ghost_scan_kernel<<<1, 1024, 0, stream>>>(mask, gidx);
    ghost_hops_kernel<<<(NGHOST + 255) / 256, 256, 0, stream>>>(gidx, first_out, first_dx,
                                                                gd1, gd2, gdx);

    for (int t = 0; t < T_IN; ++t) {
        copy_xt_kernel<<<(NN * FEAT + 255) / 256, 256, 0, stream>>>(x, t, xt);
        ghost_fuse_kernel<<<NGHOST, 160, 0, stream>>>(x, t, gidx, gd1, gd2, gdx,
                                                      Wf1, bf1, Wf2, bf2, Wg1, bg1, Wg2, bg2, xt);
        // g0 = relu(xt @ W_lin + b_lin)
        wmma_gemm_kernel<16, 32, 8, true><<<MBLK, 128, 0, stream>>>(xt, fWlin, b_lin, g);
        // GCN layer 1
        wmma_gemm_kernel<128, 128, 8, false><<<MBLK, 128, 0, stream>>>(g, fWc1, nullptr, hw);
        hipMemsetAsync(agg, 0, (size_t)NN * HID * 4, stream);
        scatter_kernel<<<NE / 8, 256, 0, stream>>>(src, dst, norm_e, hw, agg);
        gcn_finish_kernel<<<(NN * HID + 255) / 256, 256, 0, stream>>>(hw, agg, nself, bc1, g);
        // GCN layer 2
        wmma_gemm_kernel<128, 128, 8, false><<<MBLK, 128, 0, stream>>>(g, fWc2, nullptr, hw);
        hipMemsetAsync(agg, 0, (size_t)NN * HID * 4, stream);
        scatter_kernel<<<NE / 8, 256, 0, stream>>>(src, dst, norm_e, hw, agg);
        gcn_finish_kernel<<<(NN * HID + 255) / 256, 256, 0, stream>>>(hw, agg, nself, bc2, g);
        // fused GRU (updates h in place)
        gru_fused_kernel<<<MBLK, 128, 0, stream>>>(g, fWih, fWhh, b_ih, b_hh, h);
    }
}
